// PointToGrid_21801253994707
// MI455X (gfx1250) — compile-verified, matrix operand order checked by reading, not run on hardware
//
#include <hip/hip_runtime.h>
#include <math.h>

typedef _Float16 f16;
typedef _Float16 v8h  __attribute__((ext_vector_type(8)));
typedef _Float16 v16h __attribute__((ext_vector_type(16)));
typedef float    v8f  __attribute__((ext_vector_type(8)));

#define HW_GRID     65536     // 256*256
#define C_OUT       64
#define C_IN        16
#define KPAD        224       // 208 padded to 7*32
#define PTS_PER_BLK 128
#define THREADS     256
#define PI_F        3.14159265358979f

// Workspace layout (bytes)
static constexpr size_t WS_SUMS   = 0;                         // f32 [65536][64] = 16 MB
static constexpr size_t WS_COUNTS = 16777216;                  // f32 [65536]    = 256 KB
static constexpr size_t WS_W1T    = 17039360;                  // f16 [64][224]  = 28 KB (N-major, K-padded)
static constexpr size_t WS_W2T    = 17068032;                  // f16 [64][64]   = 8 KB  (N-major)

// ---------------------------------------------------------------------------
// Kernel 0: zero accumulators, convert weights to f16 N-major layout
// ---------------------------------------------------------------------------
__global__ void k0_init(const float* __restrict__ W1, const float* __restrict__ W2,
                        float* __restrict__ sums, float* __restrict__ counts,
                        f16* __restrict__ w1t, f16* __restrict__ w2t)
{
    int i = blockIdx.x * blockDim.x + threadIdx.x;
    int stride = gridDim.x * blockDim.x;
    for (int idx = i; idx < HW_GRID * C_OUT; idx += stride) sums[idx] = 0.0f;
    for (int idx = i; idx < HW_GRID; idx += stride) counts[idx] = 0.0f;
    for (int idx = i; idx < C_OUT * KPAD; idx += stride) {
        int n = idx / KPAD, k = idx - n * KPAD;
        float v = (k < 208) ? W1[k * C_OUT + n] : 0.0f;
        w1t[idx] = (f16)v;
    }
    for (int idx = i; idx < C_OUT * C_OUT; idx += stride) {
        int n = idx >> 6, k = idx & 63;
        w2t[idx] = (f16)W2[k * C_OUT + n];
    }
}

// Assemble a 16x32 f16 WMMA fragment for this lane from two 16B-aligned groups.
__device__ inline v16h load_frag(const f16* lo)
{
    union { v16h v; v8h h[2]; } u;
    u.h[0] = *(const v8h*)(lo);
    u.h[1] = *(const v8h*)(lo + 16);
    return u.v;
}

// ---------------------------------------------------------------------------
// Kernel 1: fused PE -> GEMM1 -> LN+GELU -> GEMM2 -> atomic scatter
// ---------------------------------------------------------------------------
__global__ __launch_bounds__(THREADS)
void k1_mlp_scatter(const float* __restrict__ pos, const float* __restrict__ features,
                    const float* __restrict__ b1, const float* __restrict__ ln_g,
                    const float* __restrict__ ln_b, const float* __restrict__ b2,
                    const f16* __restrict__ w1t, const f16* __restrict__ w2t,
                    const int* __restrict__ axis1p, const int* __restrict__ axis2p,
                    float* __restrict__ sums, float* __restrict__ counts, int Npts)
{
    __shared__ f16 combA[PTS_PER_BLK * KPAD];   // 57344 B; per-wave slice reused for h / h16
    __shared__ int flatA[PTS_PER_BLK];

    const int tid = threadIdx.x;
    const int p0  = blockIdx.x * PTS_PER_BLK;

    // ---- Stage 1: features | sinusoidal PE -> f16 LDS tile; grid flat index ----
    for (int idx = tid; idx < PTS_PER_BLK * KPAD; idx += THREADS) {
        int p = idx / KPAD, j = idx - p * KPAD;
        int gp = p0 + p;
        float v = 0.0f;
        if (gp < Npts) {
            if (j < C_IN) {
                v = features[gp * C_IN + j];
            } else if (j < 208) {
                int e   = j - C_IN;
                int dim = e >> 6;          // spatial dim 0..2
                int rem = e & 63;          // sin block [0,32), cos block [32,64)
                int f   = rem & 31;        // frequency index
                float x   = pos[gp * 3 + dim];
                float ang = x * 0.5f * exp2f((float)f * 0.03125f) * PI_F;
                v = (rem & 32) ? __cosf(ang) : __sinf(ang);
            }
        }
        combA[idx] = (f16)v;
    }
    if (tid < PTS_PER_BLK) {
        int gp = p0 + tid;
        int fl = -1;
        if (gp < Npts) {
            int a1 = axis1p[0], a2 = axis2p[0];
            float q1 = pos[gp * 3 + a1], q2 = pos[gp * 3 + a2];
            float n1 = fminf(fmaxf((q1 + 1.0f) * 0.5f, 0.0f), 1.0f);
            float n2 = fminf(fmaxf((q2 + 1.0f) * 0.5f, 0.0f), 1.0f);
            int i1 = min(max((int)floorf(n1 * 256.0f), 0), 255);
            int i2 = min(max((int)floorf(n2 * 256.0f), 0), 255);
            fl = i1 * 256 + i2;
        }
        flatA[tid] = fl;
    }
    __syncthreads();

    const int wave    = tid >> 5;
    const int lane    = tid & 31;
    const int halfSel = lane >> 4;      // 0: K lo-groups, 1: K hi-groups
    const int l16     = lane & 15;
    const int koff    = halfSel * 8;    // lane's K sub-offset inside a 32-chunk

    f16*   waveBase = &combA[wave * 16 * KPAD];        // 7168 B private slice
    float* hf       = (float*)waveBase;                // [16][64] f32   (0..4095)
    f16*   h16      = (f16*)((char*)waveBase + 4608);  // [16][64] f16   (4608..6655)

    // ---- Stage 2: GEMM1  [16 x 224] @ [224 x 64] -> f32 acc ----
    v8f acc[4] = {};
    const f16* arow = waveBase + l16 * KPAD;
    #pragma unroll
    for (int kc = 0; kc < 7; ++kc) {
        v16h a = load_frag(arow + kc * 32 + koff);
        #pragma unroll
        for (int t = 0; t < 4; ++t) {
            v16h b = load_frag(w1t + (t * 16 + l16) * KPAD + kc * 32 + koff);
            acc[t] = __builtin_amdgcn_wmma_f32_16x16x32_f16(
                false, a, false, b, (short)0, acc[t], false, false);
        }
    }

    // bias + spill h to LDS (own slice; A tile fully consumed)
    #pragma unroll
    for (int t = 0; t < 4; ++t) {
        float bv = b1[t * 16 + l16];
        #pragma unroll
        for (int r = 0; r < 8; ++r)
            hf[(r + 8 * halfSel) * C_OUT + (t * 16 + l16)] = acc[t][r] + bv;
    }
    __syncthreads();

    // ---- Stage 3: LayerNorm + exact GELU, repack f16 (2 lanes / point) ----
    {
        const int m  = lane >> 1;
        const int c0 = (lane & 1) * 32;
        const float* row = hf + m * C_OUT;
        float vals[32];
        float s = 0.0f, ss = 0.0f;
        #pragma unroll
        for (int i = 0; i < 32; ++i) {
            float v = row[c0 + i];
            vals[i] = v; s += v; ss += v * v;
        }
        s  += __shfl_xor(s, 1, 32);
        ss += __shfl_xor(ss, 1, 32);
        float mu   = s * (1.0f / 64.0f);
        float var  = ss * (1.0f / 64.0f) - mu * mu;
        float rstd = rsqrtf(var + 1e-5f);
        #pragma unroll
        for (int i = 0; i < 32; ++i) {
            int c = c0 + i;
            float xn = (vals[i] - mu) * rstd * ln_g[c] + ln_b[c];
            float g  = 0.5f * xn * (1.0f + erff(xn * 0.70710678f));
            h16[m * C_OUT + c] = (f16)g;
        }
    }
    __syncthreads();

    // ---- Stage 4: GEMM2  [16 x 64] @ [64 x 64] ----
    v8f acc2[4] = {};
    const f16* a2row = h16 + l16 * C_OUT;
    #pragma unroll
    for (int kc = 0; kc < 2; ++kc) {
        v16h a = load_frag(a2row + kc * 32 + koff);
        #pragma unroll
        for (int t = 0; t < 4; ++t) {
            v16h b = load_frag(w2t + (t * 16 + l16) * C_OUT + kc * 32 + koff);
            acc2[t] = __builtin_amdgcn_wmma_f32_16x16x32_f16(
                false, a, false, b, (short)0, acc2[t], false, false);
        }
    }

    // ---- Stage 5: scatter-add into L2-resident accumulator ----
    #pragma unroll
    for (int t = 0; t < 4; ++t) {
        int n = t * 16 + l16;
        float bv = b2[n];
        #pragma unroll
        for (int r = 0; r < 8; ++r) {
            int m  = r + 8 * halfSel;
            int fl = flatA[wave * 16 + m];
            if (fl >= 0)
                atomicAdd(&sums[fl * C_OUT + n], acc2[t][r] + bv);
        }
    }
    if (lane < 16) {
        int fl = flatA[wave * 16 + lane];
        if (fl >= 0) atomicAdd(&counts[fl], 1.0f);
    }
}

// ---------------------------------------------------------------------------
// Kernel 2: mean + transpose to [1, C, H, W]
// ---------------------------------------------------------------------------
__global__ void k2_finalize(const float* __restrict__ sums, const float* __restrict__ counts,
                            float* __restrict__ out)
{
    int idx = blockIdx.x * blockDim.x + threadIdx.x;
    if (idx >= C_OUT * HW_GRID) return;
    int c   = idx >> 16;
    int pix = idx & (HW_GRID - 1);
    out[idx] = sums[pix * C_OUT + c] / fmaxf(counts[pix], 1.0f);
}

// ---------------------------------------------------------------------------
extern "C" void kernel_launch(void* const* d_in, const int* in_sizes, int n_in,
                              void* d_out, int out_size, void* d_ws, size_t ws_size,
                              hipStream_t stream)
{
    const float* pos      = (const float*)d_in[0];
    const float* features = (const float*)d_in[1];
    const float* W1       = (const float*)d_in[2];
    const float* b1       = (const float*)d_in[3];
    const float* ln_g     = (const float*)d_in[4];
    const float* ln_b     = (const float*)d_in[5];
    const float* W2       = (const float*)d_in[6];
    const float* b2       = (const float*)d_in[7];
    const int*   axis1    = (const int*)d_in[8];
    const int*   axis2    = (const int*)d_in[9];
    const int    Npts     = in_sizes[0] / 3;

    float* sums   = (float*)((char*)d_ws + WS_SUMS);
    float* counts = (float*)((char*)d_ws + WS_COUNTS);
    f16*   w1t    = (f16*)((char*)d_ws + WS_W1T);
    f16*   w2t    = (f16*)((char*)d_ws + WS_W2T);

    k0_init<<<2048, 256, 0, stream>>>(W1, W2, sums, counts, w1t, w2t);

    int nblk = (Npts + PTS_PER_BLK - 1) / PTS_PER_BLK;
    k1_mlp_scatter<<<nblk, THREADS, 0, stream>>>(
        pos, features, b1, ln_g, ln_b, b2, w1t, w2t, axis1, axis2, sums, counts, Npts);

    k2_finalize<<<(C_OUT * HW_GRID + 255) / 256, 256, 0, stream>>>(
        sums, counts, (float*)d_out);
}
